// OwlViTForObjectDetection_36876589204015
// MI455X (gfx1250) — matrix operand order repaired and checked by reading, not compiled, not run on hardware
//
#include <hip/hip_runtime.h>
#include <cstddef>
#include <cstdint>

// ---------------------------------------------------------------------------
// Types for CDNA5 WMMA (gfx1250, wave32)
// ---------------------------------------------------------------------------
typedef __attribute__((ext_vector_type(16))) _Float16 v16h;
typedef __attribute__((ext_vector_type(8)))  float    v8f;

#define BM 128
#define BN 64
#define BKK 32
#define GEMM_THREADS 128
#define LDH 40  // padded LDS row stride in halfs: 80 bytes, multiple of 16

union FragU {
  struct { uint4 lo, hi; } u;
  v16h v;
};

union PackU {
  _Float16 h[16];
  uint4 q[2];
};

__device__ __forceinline__ float act_apply(float x, int act) {
  if (act == 1) {                       // quick-gelu: x * sigmoid(1.702 x)
    return x / (1.0f + __expf(-1.702f * x));
  } else if (act == 2) {                // exact gelu (erf)
    return 0.5f * x * (1.0f + erff(x * 0.70710678118654752f));
  }
  return x;
}

// ---------------------------------------------------------------------------
// Universal batched GEMM:  C = act( alpha*(A·B + bias) + res )
//   A: [M,K] row-major (lda), per batch offset outer*sAo + inner*sAi
//   B: transB=1 -> [N,K] row-major (ldb)  (PyTorch Linear weight layout)
//      transB=0 -> [K,N] row-major (ldb)
//   batch index bz decomposes as outer = bz/H, inner = bz%H
//   Tile: 128M x 64N x 32K, 4 waves, each wave = 32M x 64N (2x4 fragments)
// ---------------------------------------------------------------------------
__global__ __launch_bounds__(GEMM_THREADS)
void gemm_wmma_kernel(const float* __restrict__ A, const float* __restrict__ Bm,
                      const float* __restrict__ bias, const float* __restrict__ res,
                      float* __restrict__ C,
                      int M, int N, int K, int lda, int ldb, int ldc,
                      long long sAo, long long sAi, long long sBo, long long sBi,
                      long long sCo, long long sCi, int H,
                      int transB, float alpha, int act)
{
  __shared__ __align__(16) _Float16 As[BM][LDH];
  __shared__ __align__(16) _Float16 Bs[BN][LDH];

  const int tid   = threadIdx.x;
  const int lane  = tid & 31;
  const int wv    = tid >> 5;
  const int waveM = wv * 32;
  const int lm    = lane & 15;   // fragment row/col within 16
  const int lh    = lane >> 4;   // half-wave select

  const int bz    = blockIdx.z;
  const int outer = bz / H;
  const int inner = bz % H;
  const float* Ap = A  + (long long)outer * sAo + (long long)inner * sAi;
  const float* Bp = Bm + (long long)outer * sBo + (long long)inner * sBi;
  float*       Cp = C  + (long long)outer * sCo + (long long)inner * sCi;
  const float* Rp = res ? res + (long long)outer * sCo + (long long)inner * sCi : nullptr;

  const int gm0 = blockIdx.y * BM;
  const int gn0 = blockIdx.x * BN;

  // ---- per-thread staging geometry (contiguous 16-wide chunks) ----
  const int c0  = (tid & 1) * 16;      // k-chunk base 0 or 16
  // transB=1: B staged like A with N rows
  const int rB  = tid >> 1;
  const bool bRowOk = (gn0 + rB) < N;
  const long long bRowIdx = bRowOk ? (long long)(gn0 + rB) : 0;
  // transB=0: thread covers k = tid>>2 (0..31), n chunk (tid&3)*16
  const int kB  = tid >> 2;
  const int n0B = (tid & 3) * 16;
  const int nRem = N - gn0;            // > 0 by grid construction

  v8f zero = {};
  v8f acc[2][4];
#pragma unroll
  for (int fr = 0; fr < 2; ++fr)
#pragma unroll
    for (int fc = 0; fc < 4; ++fc) acc[fr][fc] = zero;

  for (int k0 = 0; k0 < K; k0 += BKK) {
    const int kl = K - k0;             // remaining K this step (>0)

    // ---- stage A tile [128 x 32] as f16 (two 16-chunks per thread) ----
#pragma unroll
    for (int ch = 0; ch < 2; ++ch) {
      const int rA = (tid + ch * GEMM_THREADS) >> 1;
      const bool aRowOk = (gm0 + rA) < M;
      const long long aRowIdx = aRowOk ? (long long)(gm0 + rA) : 0;
      PackU pk;
      const float* rp = Ap + aRowIdx * lda + k0;
      if (aRowOk && (c0 + 16 <= kl)) {
#pragma unroll
        for (int j = 0; j < 16; ++j) pk.h[j] = (_Float16)rp[c0 + j];
      } else {
#pragma unroll
        for (int j = 0; j < 16; ++j) {
          int gk = c0 + j;
          float v = rp[gk < kl ? gk : 0];
          pk.h[j] = (_Float16)((aRowOk && gk < kl) ? v : 0.0f);
        }
      }
      uint4* dst = reinterpret_cast<uint4*>(&As[rA][c0]);
      dst[0] = pk.q[0];
      dst[1] = pk.q[1];
    }

    // ---- stage B tile [64n x 32k] as f16 ----
    if (transB) {
      PackU pk;
      const float* rp = Bp + bRowIdx * ldb + k0;
      if (bRowOk && (c0 + 16 <= kl)) {
#pragma unroll
        for (int j = 0; j < 16; ++j) pk.h[j] = (_Float16)rp[c0 + j];
      } else {
#pragma unroll
        for (int j = 0; j < 16; ++j) {
          int gk = c0 + j;
          float v = rp[gk < kl ? gk : 0];
          pk.h[j] = (_Float16)((bRowOk && gk < kl) ? v : 0.0f);
        }
      }
      uint4* dst = reinterpret_cast<uint4*>(&Bs[rB][c0]);
      dst[0] = pk.q[0];
      dst[1] = pk.q[1];
    } else {
      const bool kOk = kB < kl;
      const float* rp = Bp + (long long)(k0 + (kOk ? kB : 0)) * ldb + gn0;
      _Float16 h[16];
      if (kOk && (n0B + 16 <= nRem)) {
#pragma unroll
        for (int j = 0; j < 16; ++j) h[j] = (_Float16)rp[n0B + j];
      } else {
#pragma unroll
        for (int j = 0; j < 16; ++j) {
          int gn = n0B + j;
          float v = rp[gn < nRem ? gn : (nRem - 1)];
          h[j] = (_Float16)((kOk && gn < nRem) ? v : 0.0f);
        }
      }
#pragma unroll
      for (int j = 0; j < 16; ++j) Bs[n0B + j][kB] = h[j];
    }
    __syncthreads();

    // ---- fragment loads in contiguous k order (same for A and B) ----
    FragU af[2], bf[4];
#pragma unroll
    for (int fr = 0; fr < 2; ++fr) {
      int m = waveM + fr * 16 + lm;
      // element i holds K = 16*lh + i
      af[fr].u.lo = *reinterpret_cast<const uint4*>(&As[m][lh * 16]);
      af[fr].u.hi = *reinterpret_cast<const uint4*>(&As[m][lh * 16 + 8]);
    }
#pragma unroll
    for (int fc = 0; fc < 4; ++fc) {
      int n = fc * 16 + lm;
      // element i holds K = 16*lh + i
      const uint4* p = reinterpret_cast<const uint4*>(&Bs[n][lh * 16]);
      bf[fc].u.lo = p[0];
      bf[fc].u.hi = p[1];
    }
#pragma unroll
    for (int fr = 0; fr < 2; ++fr)
#pragma unroll
      for (int fc = 0; fc < 4; ++fc)
        acc[fr][fc] = __builtin_amdgcn_wmma_f32_16x16x32_f16(
            false, af[fr].v, false, bf[fc].v, (short)0, acc[fr][fc], false, false);
    __syncthreads();
  }

  // ---- epilogue: C/D layout m = r + 8*lh, n = lm ----
#pragma unroll
  for (int fc = 0; fc < 4; ++fc) {
    int gn = gn0 + fc * 16 + lm;
    if (gn >= N) continue;
    float bn = bias ? bias[gn] : 0.0f;
#pragma unroll
    for (int fr = 0; fr < 2; ++fr) {
#pragma unroll
      for (int r = 0; r < 8; ++r) {
        int gm = gm0 + waveM + fr * 16 + lh * 8 + r;
        if (gm < M) {
          float v = alpha * (acc[fr][fc][r] + bn);
          if (Rp) v += Rp[(long long)gm * ldc + gn];
          Cp[(long long)gm * ldc + gn] = act_apply(v, act);
        }
      }
    }
  }
}

// ---------------------------------------------------------------------------
// LayerNorm: one 256-thread block per row (safe in-place)
// ---------------------------------------------------------------------------
__global__ __launch_bounds__(256)
void ln_kernel(const float* __restrict__ x, const float* __restrict__ w,
               const float* __restrict__ b, float* __restrict__ y, int cols)
{
  __shared__ float s0[256], s1[256];
  long long row = blockIdx.x;
  const float* xr = x + row * cols;
  float sum = 0.f, sq = 0.f;
  for (int c = threadIdx.x; c < cols; c += 256) { float v = xr[c]; sum += v; sq += v * v; }
  s0[threadIdx.x] = sum; s1[threadIdx.x] = sq;
  __syncthreads();
  for (int st = 128; st > 0; st >>= 1) {
    if (threadIdx.x < st) { s0[threadIdx.x] += s0[threadIdx.x + st]; s1[threadIdx.x] += s1[threadIdx.x + st]; }
    __syncthreads();
  }
  float mu  = s0[0] / cols;
  float var = s1[0] / cols - mu * mu;
  float rstd = rsqrtf(var + 1e-5f);
  float* yr = y + row * cols;
  for (int c = threadIdx.x; c < cols; c += 256)
    yr[c] = (xr[c] - mu) * rstd * w[c] + b[c];
}

// ---------------------------------------------------------------------------
// Row softmax (in place): one 128-thread block per row
// ---------------------------------------------------------------------------
__global__ __launch_bounds__(128)
void softmax_kernel(float* __restrict__ x, int cols)
{
  __shared__ float red[128];
  long long row = blockIdx.x;
  float* xr = x + row * cols;
  float m = -3.402823466e38f;
  for (int c = threadIdx.x; c < cols; c += 128) m = fmaxf(m, xr[c]);
  red[threadIdx.x] = m; __syncthreads();
  for (int st = 64; st > 0; st >>= 1) {
    if (threadIdx.x < st) red[threadIdx.x] = fmaxf(red[threadIdx.x], red[threadIdx.x + st]);
    __syncthreads();
  }
  float mx = red[0]; __syncthreads();
  float s = 0.f;
  for (int c = threadIdx.x; c < cols; c += 128) { float e = __expf(xr[c] - mx); xr[c] = e; s += e; }
  red[threadIdx.x] = s; __syncthreads();
  for (int st = 64; st > 0; st >>= 1) {
    if (threadIdx.x < st) red[threadIdx.x] += red[threadIdx.x + st];
    __syncthreads();
  }
  float inv = 1.0f / red[0];
  for (int c = threadIdx.x; c < cols; c += 128) xr[c] *= inv;
}

// ---------------------------------------------------------------------------
// Row L2 normalize (in place): x /= (||x|| + 1e-6)
// ---------------------------------------------------------------------------
__global__ __launch_bounds__(256)
void l2norm_kernel(float* __restrict__ x, int cols)
{
  __shared__ float red[256];
  long long row = blockIdx.x;
  float* xr = x + row * cols;
  float s = 0.f;
  for (int c = threadIdx.x; c < cols; c += 256) { float v = xr[c]; s += v * v; }
  red[threadIdx.x] = s; __syncthreads();
  for (int st = 128; st > 0; st >>= 1) {
    if (threadIdx.x < st) red[threadIdx.x] += red[threadIdx.x + st];
    __syncthreads();
  }
  float inv = 1.0f / (sqrtf(red[0]) + 1e-6f);
  for (int c = threadIdx.x; c < cols; c += 256) xr[c] *= inv;
}

// ---------------------------------------------------------------------------
// Elementwise / gather kernels
// ---------------------------------------------------------------------------
__global__ void patchify_kernel(const float* __restrict__ px, float* __restrict__ out, int total)
{
  int idx = blockIdx.x * 256 + threadIdx.x;
  if (idx >= total) return;
  int m = idx / 3072, k = idx % 3072;
  int b = m / 576, pp = m % 576;
  int hh = pp / 24, ww = pp % 24;
  int c = k / 1024, rem = k % 1024;
  int p = rem / 32, q = rem % 32;
  out[idx] = px[(((long long)b * 3 + c) * 768 + hh * 32 + p) * 768 + ww * 32 + q];
}

__global__ void vis_embed_kernel(const float* __restrict__ pemb, const float* __restrict__ cls,
                                 const float* __restrict__ pos, float* __restrict__ x, int total)
{
  int idx = blockIdx.x * 256 + threadIdx.x;
  if (idx >= total) return;
  int row = idx / 768, c = idx % 768;
  int b = row / 577, t = row % 577;
  float v = (t == 0) ? cls[c] : pemb[((long long)b * 576 + (t - 1)) * 768 + c];
  x[idx] = v + pos[(long long)t * 768 + c];
}

__global__ void txt_embed_kernel(const int* __restrict__ ids, const float* __restrict__ tok,
                                 const float* __restrict__ pos, float* __restrict__ x, int total)
{
  int idx = blockIdx.x * 256 + threadIdx.x;
  if (idx >= total) return;
  int row = idx / 512, c = idx % 512;
  int j = row % 16;
  x[idx] = tok[(long long)ids[row] * 512 + c] + pos[(long long)j * 512 + c];
}

__global__ void merge_kernel(const float* __restrict__ img, float* __restrict__ feats, int total)
{
  int idx = blockIdx.x * 256 + threadIdx.x;
  if (idx >= total) return;
  int row = idx / 768, c = idx % 768;
  int b = row / 576, p = row % 576;
  feats[idx] = img[((long long)b * 577 + 1 + p) * 768 + c] * img[(long long)b * 577 * 768 + c];
}

__global__ __launch_bounds__(128)
void pool_kernel(const int* __restrict__ ids, const float* __restrict__ t, float* __restrict__ pooled)
{
  __shared__ int jm;
  int s = blockIdx.x;
  if (threadIdx.x == 0) {
    int best = ids[s * 16], bj = 0;
    for (int j = 1; j < 16; ++j) { int v = ids[s * 16 + j]; if (v > best) { best = v; bj = j; } }
    jm = bj;
  }
  __syncthreads();
  for (int c = threadIdx.x; c < 512; c += 128)
    pooled[(long long)s * 512 + c] = t[((long long)s * 16 + jm) * 512 + c];
}

__global__ __launch_bounds__(256)
void shift_scale_kernel(const float* __restrict__ feats,
                        const float* __restrict__ sw, const float* __restrict__ sb,
                        const float* __restrict__ cw, const float* __restrict__ cb,
                        float* __restrict__ shiftv, float* __restrict__ scalev)
{
  __shared__ float r0[256], r1[256];
  long long row = blockIdx.x;
  const float* fr = feats + row * 768;
  float a = 0.f, b2 = 0.f;
  for (int c = threadIdx.x; c < 768; c += 256) { float f = fr[c]; a += f * sw[c]; b2 += f * cw[c]; }
  r0[threadIdx.x] = a; r1[threadIdx.x] = b2; __syncthreads();
  for (int st = 128; st > 0; st >>= 1) {
    if (threadIdx.x < st) { r0[threadIdx.x] += r0[threadIdx.x + st]; r1[threadIdx.x] += r1[threadIdx.x + st]; }
    __syncthreads();
  }
  if (threadIdx.x == 0) { shiftv[row] = r0[0] + sb[0]; scalev[row] = r1[0] + cb[0]; }
}

__global__ void logits_combine_kernel(const float* __restrict__ logits,
                                      const float* __restrict__ shiftv,
                                      const float* __restrict__ scalev,
                                      float* __restrict__ out, int total)
{
  int idx = blockIdx.x * 256 + threadIdx.x;
  if (idx >= total) return;
  int row = idx / 16;
  float sh = shiftv[row];
  float sr = scalev[row];
  float sc = sr > 0.f ? sr + 1.f : __expf(sr);   // elu(x)+1
  out[idx] = (logits[idx] + sh) * sc;
}

__global__ __launch_bounds__(256)
void box_out_kernel(const float* __restrict__ bh, const float* __restrict__ w,
                    const float* __restrict__ bvec, float* __restrict__ out)
{
  __shared__ float red[4][256];
  long long row = blockIdx.x;    // 0..4607
  const float* xr = bh + row * 768;
  float p0 = 0.f, p1 = 0.f, p2 = 0.f, p3 = 0.f;
  for (int c = threadIdx.x; c < 768; c += 256) {
    float v = xr[c];
    p0 += v * w[c]; p1 += v * w[768 + c]; p2 += v * w[1536 + c]; p3 += v * w[2304 + c];
  }
  red[0][threadIdx.x] = p0; red[1][threadIdx.x] = p1;
  red[2][threadIdx.x] = p2; red[3][threadIdx.x] = p3;
  __syncthreads();
  for (int st = 128; st > 0; st >>= 1) {
    if (threadIdx.x < st)
      for (int j = 0; j < 4; ++j) red[j][threadIdx.x] += red[j][threadIdx.x + st];
    __syncthreads();
  }
  if (threadIdx.x < 4) {
    int p = (int)(row % 576);
    int rr = p / 24, cc = p % 24;
    float xx = (cc + 1) / 24.0f, yy = (rr + 1) / 24.0f;
    float bias;
    if (threadIdx.x == 0)       bias = logf(xx + 1e-4f) - logf(1.f - xx + 1e-4f);
    else if (threadIdx.x == 1)  bias = logf(yy + 1e-4f) - logf(1.f - yy + 1e-4f);
    else { const float s = 1.f / 24.f; bias = logf(s + 1e-4f) - logf(1.f - s + 1e-4f); }
    float v = red[threadIdx.x][0] + bvec[threadIdx.x] + bias;
    out[row * 4 + threadIdx.x] = 1.f / (1.f + __expf(-v));
  }
}

// ---------------------------------------------------------------------------
// Host-side orchestration
// ---------------------------------------------------------------------------
static inline int cdiv(int a, int b) { return (a + b - 1) / b; }

static void gemm(hipStream_t st, const float* A, const float* B, const float* bias,
                 const float* res, float* C, int M, int N, int K,
                 int lda, int ldb, int ldc, int batches, int H,
                 long long sAo, long long sAi, long long sBo, long long sBi,
                 long long sCo, long long sCi, int transB, float alpha, int act)
{
  dim3 g(cdiv(N, BN), cdiv(M, BM), batches), blk(GEMM_THREADS);
  gemm_wmma_kernel<<<g, blk, 0, st>>>(A, B, bias, res, C, M, N, K, lda, ldb, ldc,
                                      sAo, sAi, sBo, sBi, sCo, sCi, H, transB, alpha, act);
}

static void gemm1(hipStream_t st, const float* A, const float* B, const float* bias,
                  const float* res, float* C, int M, int N, int K,
                  int lda, int ldb, int ldc, int transB, float alpha, int act)
{
  gemm(st, A, B, bias, res, C, M, N, K, lda, ldb, ldc, 1, 1, 0, 0, 0, 0, 0, 0, transB, alpha, act);
}

extern "C" void kernel_launch(void* const* d_in, const int* in_sizes, int n_in,
                              void* d_out, int out_size, void* d_ws, size_t ws_size,
                              hipStream_t stream)
{
  (void)in_sizes; (void)n_in; (void)out_size; (void)ws_size;

  // ---- model dims ----
  const int VROWS = 8 * 577;     // 4616
  const int TROWS = 128 * 16;    // 2048
  const int NPR   = 8 * 576;     // 4608 patch rows

  // ---- input pointers (setup_inputs dict order, recursively flattened) ----
  const float* pixel     = (const float*)d_in[0];
  const int*   ids       = (const int*)  d_in[1];
  const float* patch_w   = (const float*)d_in[2];
  const float* v_cls     = (const float*)d_in[3];
  const float* v_pos     = (const float*)d_in[4];
  const float* v_prew    = (const float*)d_in[5];
  const float* v_preb    = (const float*)d_in[6];
  const float* vW[16];  for (int i = 0; i < 16; ++i) vW[i] = (const float*)d_in[7 + i];
  const float* v_postw   = (const float*)d_in[23];
  const float* v_postb   = (const float*)d_in[24];
  const float* tok_emb   = (const float*)d_in[25];
  const float* pos_emb   = (const float*)d_in[26];
  const float* tW[16];  for (int i = 0; i < 16; ++i) tW[i] = (const float*)d_in[27 + i];
  const float* fin_w     = (const float*)d_in[43];
  const float* fin_b     = (const float*)d_in[44];
  const float* proj_w    = (const float*)d_in[45];
  const float* mrg_w     = (const float*)d_in[46];
  const float* mrg_b     = (const float*)d_in[47];
  const float* cls0_w    = (const float*)d_in[48];
  const float* cls0_b    = (const float*)d_in[49];
  const float* shf_w     = (const float*)d_in[50];
  const float* shf_b     = (const float*)d_in[51];
  const float* scl_w     = (const float*)d_in[52];
  const float* scl_b     = (const float*)d_in[53];
  const float* bx0_w     = (const float*)d_in[54];
  const float* bx0_b     = (const float*)d_in[55];
  const float* bx1_w     = (const float*)d_in[56];
  const float* bx1_b     = (const float*)d_in[57];
  const float* bx2_w     = (const float*)d_in[58];
  const float* bx2_b     = (const float*)d_in[59];

  // ---- workspace layout (floats) ----
  float* W = (float*)d_ws;
  size_t off = 0;
  auto alloc = [&](size_t n) { float* p = W + off; off += (n + 63) & ~(size_t)63; return p; };

  float* patches = alloc((size_t)NPR * 3072);
  float* vxa  = alloc((size_t)VROWS * 768);
  float* vxb  = alloc((size_t)VROWS * 768);
  float* vxn  = alloc((size_t)VROWS * 768);
  float* vq   = alloc((size_t)VROWS * 768);
  float* vk   = alloc((size_t)VROWS * 768);
  float* vv   = alloc((size_t)VROWS * 768);
  float* vctx = alloc((size_t)VROWS * 768);
  float* vsc  = alloc((size_t)96 * 577 * 577);
  float* vhb  = alloc((size_t)VROWS * 3072);
  float* feats= alloc((size_t)NPR * 768);
  float* txa  = alloc((size_t)TROWS * 512);
  float* txb  = alloc((size_t)TROWS * 512);
  float* txn  = alloc((size_t)TROWS * 512);
  float* tq   = alloc((size_t)TROWS * 512);
  float* tk   = alloc((size_t)TROWS * 512);
  float* tv   = alloc((size_t)TROWS * 512);
  float* tctx = alloc((size_t)TROWS * 512);
  float* tsc  = alloc((size_t)1024 * 256);
  float* thb  = alloc((size_t)TROWS * 2048);
  float* pooled = alloc((size_t)128 * 512);
  float* qn     = alloc((size_t)128 * 512);
  float* ice    = alloc((size_t)NPR * 512);
  float* logitsb= alloc((size_t)8 * 576 * 16);
  float* shiftv = alloc((size_t)NPR);
  float* scalev = alloc((size_t)NPR);
  float* bh1    = alloc((size_t)NPR * 768);
  float* bh2    = alloc((size_t)NPR * 768);

  float* out = (float*)d_out;

  // ===================== vision tower =====================
  {
    int total = NPR * 3072;
    patchify_kernel<<<cdiv(total, 256), 256, 0, stream>>>(pixel, patches, total);
  }
  gemm1(stream, patches, patch_w, nullptr, nullptr, vctx, NPR, 768, 3072, 3072, 3072, 768, 1, 1.0f, 0);
  {
    int total = VROWS * 768;
    vis_embed_kernel<<<cdiv(total, 256), 256, 0, stream>>>(vctx, v_cls, v_pos, vxa, total);
  }
  ln_kernel<<<VROWS, 256, 0, stream>>>(vxa, v_prew, v_preb, vxa, 768);

  const long long VSC = 577LL * 577LL;
  for (int i = 0; i < 12; ++i) {
    const float* qw = vW[0] + (size_t)i * 768 * 768;  const float* qb = vW[1] + (size_t)i * 768;
    const float* kw = vW[2] + (size_t)i * 768 * 768;  const float* kb = vW[3] + (size_t)i * 768;
    const float* vw = vW[4] + (size_t)i * 768 * 768;  const float* vb = vW[5] + (size_t)i * 768;
    const float* ow = vW[6] + (size_t)i * 768 * 768;  const float* ob = vW[7] + (size_t)i * 768;
    const float* l1w= vW[8] + (size_t)i * 768;        const float* l1b= vW[9] + (size_t)i * 768;
    const float* f1w= vW[10]+ (size_t)i * 3072 * 768; const float* f1b= vW[11]+ (size_t)i * 3072;
    const float* f2w= vW[12]+ (size_t)i * 768 * 3072; const float* f2b= vW[13]+ (size_t)i * 768;
    const float* l2w= vW[14]+ (size_t)i * 768;        const float* l2b= vW[15]+ (size_t)i * 768;

    ln_kernel<<<VROWS, 256, 0, stream>>>(vxa, l1w, l1b, vxn, 768);
    gemm1(stream, vxn, qw, qb, nullptr, vq, VROWS, 768, 768, 768, 768, 768, 1, 0.125f, 0);
    gemm1(stream, vxn, kw, kb, nullptr, vk, VROWS, 768, 768, 768, 768, 768, 1, 1.0f, 0);
    gemm1(stream, vxn, vw, vb, nullptr, vv, VROWS, 768, 768, 768, 768, 768, 1, 1.0f, 0);
    gemm(stream, vq, vk, nullptr, nullptr, vsc, 577, 577, 64, 768, 768, 577,
         96, 12, 577LL * 768, 64, 577LL * 768, 64, 12LL * VSC, VSC, 1, 1.0f, 0);
    softmax_kernel<<<96 * 577, 128, 0, stream>>>(vsc, 577);
    gemm(stream, vsc, vv, nullptr, nullptr, vctx, 577, 64, 577, 577, 768, 768,
         96, 12, 12LL * VSC, VSC, 577LL * 768, 64, 577LL * 768, 64, 0, 1.0f, 0);
    gemm1(stream, vctx, ow, ob, vxa, vxb, VROWS, 768, 768, 768, 768, 768, 1, 1.0f, 0);
    ln_kernel<<<VROWS, 256, 0, stream>>>(vxb, l2w, l2b, vxn, 768);
    gemm1(stream, vxn, f1w, f1b, nullptr, vhb, VROWS, 3072, 768, 768, 768, 3072, 1, 1.0f, 1);
    gemm1(stream, vhb, f2w, f2b, vxb, vxa, VROWS, 768, 3072, 3072, 3072, 768, 1, 1.0f, 0);
  }
  ln_kernel<<<VROWS, 256, 0, stream>>>(vxa, v_postw, v_postb, vxn, 768);
  {
    int total = NPR * 768;
    merge_kernel<<<cdiv(total, 256), 256, 0, stream>>>(vxn, feats, total);
  }
  ln_kernel<<<NPR, 256, 0, stream>>>(feats, mrg_w, mrg_b, feats, 768);

  // ===================== text tower =====================
  {
    int total = TROWS * 512;
    txt_embed_kernel<<<cdiv(total, 256), 256, 0, stream>>>(ids, tok_emb, pos_emb, txa, total);
  }
  for (int i = 0; i < 12; ++i) {
    const float* qw = tW[0] + (size_t)i * 512 * 512;  const float* qb = tW[1] + (size_t)i * 512;
    const float* kw = tW[2] + (size_t)i * 512 * 512;  const float* kb = tW[3] + (size_t)i * 512;
    const float* vw = tW[4] + (size_t)i * 512 * 512;  const float* vb = tW[5] + (size_t)i * 512;
    const float* ow = tW[6] + (size_t)i * 512 * 512;  const float* ob = tW[7] + (size_t)i * 512;
    const float* l1w= tW[8] + (size_t)i * 512;        const float* l1b= tW[9] + (size_t)i * 512;
    const float* f1w= tW[10]+ (size_t)i * 2048 * 512; const float* f1b= tW[11]+ (size_t)i * 2048;
    const float* f2w= tW[12]+ (size_t)i * 512 * 2048; const float* f2b= tW[13]+ (size_t)i * 512;
    const float* l2w= tW[14]+ (size_t)i * 512;        const float* l2b= tW[15]+ (size_t)i * 512;

    ln_kernel<<<TROWS, 256, 0, stream>>>(txa, l1w, l1b, txn, 512);
    gemm1(stream, txn, qw, qb, nullptr, tq, TROWS, 512, 512, 512, 512, 512, 1, 0.125f, 0);
    gemm1(stream, txn, kw, kb, nullptr, tk, TROWS, 512, 512, 512, 512, 512, 1, 1.0f, 0);
    gemm1(stream, txn, vw, vb, nullptr, tv, TROWS, 512, 512, 512, 512, 512, 1, 1.0f, 0);
    gemm(stream, tq, tk, nullptr, nullptr, tsc, 16, 16, 64, 512, 512, 16,
         1024, 8, 16LL * 512, 64, 16LL * 512, 64, 8LL * 256, 256, 1, 1.0f, 0);
    softmax_kernel<<<1024 * 16, 128, 0, stream>>>(tsc, 16);
    gemm(stream, tsc, tv, nullptr, nullptr, tctx, 16, 64, 16, 16, 512, 512,
         1024, 8, 8LL * 256, 256, 16LL * 512, 64, 16LL * 512, 64, 0, 1.0f, 0);
    gemm1(stream, tctx, ow, ob, txa, txb, TROWS, 512, 512, 512, 512, 512, 1, 1.0f, 0);
    ln_kernel<<<TROWS, 256, 0, stream>>>(txb, l2w, l2b, txn, 512);
    gemm1(stream, txn, f1w, f1b, nullptr, thb, TROWS, 2048, 512, 512, 512, 2048, 1, 1.0f, 1);
    gemm1(stream, thb, f2w, f2b, txb, txa, TROWS, 512, 2048, 2048, 2048, 512, 1, 1.0f, 0);
  }
  ln_kernel<<<TROWS, 256, 0, stream>>>(txa, fin_w, fin_b, txa, 512);
  pool_kernel<<<128, 128, 0, stream>>>(ids, txa, pooled);
  gemm1(stream, pooled, proj_w, nullptr, nullptr, qn, 128, 512, 512, 512, 512, 512, 1, 1.0f, 0);
  l2norm_kernel<<<128, 256, 0, stream>>>(qn, 512);

  // ===================== heads =====================
  gemm1(stream, feats, cls0_w, cls0_b, nullptr, ice, NPR, 512, 768, 768, 768, 512, 1, 1.0f, 0);
  l2norm_kernel<<<NPR, 256, 0, stream>>>(ice, 512);
  gemm(stream, ice, qn, nullptr, nullptr, logitsb, 576, 16, 512, 512, 512, 16,
       8, 1, 576LL * 512, 0, 16LL * 512, 0, 576LL * 16, 0, 1, 1.0f, 0);
  shift_scale_kernel<<<NPR, 256, 0, stream>>>(feats, shf_w, shf_b, scl_w, scl_b, shiftv, scalev);
  {
    int total = 8 * 576 * 16;
    logits_combine_kernel<<<cdiv(total, 256), 256, 0, stream>>>(logitsb, shiftv, scalev, out, total);
  }
  gemm1(stream, feats, bx0_w, bx0_b, nullptr, bh1, NPR, 768, 768, 768, 768, 768, 1, 1.0f, 2);
  gemm1(stream, bh1,   bx1_w, bx1_b, nullptr, bh2, NPR, 768, 768, 768, 768, 768, 1, 1.0f, 2);
  box_out_kernel<<<NPR, 256, 0, stream>>>(bh2, bx2_w, bx2_b, out + 8 * 576 * 16);
}
